// InfoNCELoss_40321152974998
// MI455X (gfx1250) — matrix operand order repaired and checked by reading, not compiled, not run on hardware
//
#include <hip/hip_runtime.h>
#include <hip/hip_bf16.h>

typedef __attribute__((ext_vector_type(16))) _Float16 v16h;
typedef __attribute__((ext_vector_type(8)))  _Float16 v8h;
typedef __attribute__((ext_vector_type(4)))  _Float16 v4h;
typedef __attribute__((ext_vector_type(8)))  float    v8f;

#define NROWS 4096
#define DDIM  1024
#define TWON  8192
#define INVT  14.285714285714286f   // 1/TEMP, also the fixed logsumexp shift

constexpr int ROWS_WG  = 128;        // 8 waves * 16 rows
constexpr int COLS_BLK = 64;         // 4 WMMA tiles per wave
constexpr int KCH      = 128;        // B K-chunk staged in LDS (double buffered)
constexpr int ASTR     = DDIM + 8;   // resident A stride: 2064B, conflict-free
constexpr int BST      = KCH + 8;    // B stride: 272B, conflict-free
constexpr int COL_SPLITS = 4;
constexpr int JBLKS    = TWON / COLS_BLK / COL_SPLITS;  // 32
constexpr int KCHUNKS  = DDIM / KCH;                    // 8
constexpr int A_TOT    = ROWS_WG * ASTR;                // 132096 halves (258KB)
constexpr int B_HALVES = COLS_BLK * BST;                // 8704 halves / buffer

// ---------------------------------------------------------------------------
// CDNA5 async global->LDS copy (16B per lane), tracked by ASYNCcnt.
// VDST VGPR holds the LDS byte offset (generic LDS ptr truncates to offset).
// ---------------------------------------------------------------------------
__device__ __forceinline__ void async_cp16(const _Float16* g, const _Float16* l) {
  unsigned ldsoff = (unsigned)(unsigned long long)l;
  asm volatile("global_load_async_to_lds_b128 %0, %1, off"
               :: "v"(ldsoff), "v"(g)
               : "memory");
}

__device__ __forceinline__ void wait_async_le4() {
#if defined(__has_builtin) && __has_builtin(__builtin_amdgcn_s_wait_asynccnt)
  __builtin_amdgcn_s_wait_asynccnt(4);
#else
  asm volatile("s_wait_asynccnt 4" ::: "memory");
#endif
}
__device__ __forceinline__ void wait_async_le0() {
#if defined(__has_builtin) && __has_builtin(__builtin_amdgcn_s_wait_asynccnt)
  __builtin_amdgcn_s_wait_asynccnt(0);
#else
  asm volatile("s_wait_asynccnt 0" ::: "memory");
#endif
}

// ---------------------------------------------------------------------------
// prep: fp32 -> f16 conversion of Z=[z1;z2], zero rowsum + out (each call)
// ---------------------------------------------------------------------------
__global__ __launch_bounds__(256) void prep_kernel(const float* __restrict__ z1,
                                                   const float* __restrict__ z2,
                                                   _Float16* __restrict__ zh,
                                                   float* __restrict__ rowsum,
                                                   float* __restrict__ out) {
  unsigned g   = blockIdx.x * 256u + threadIdx.x;
  unsigned off = g * 4u;
  float4 v;
  if (off < (unsigned)NROWS * DDIM) v = *(const float4*)(z1 + off);
  else                              v = *(const float4*)(z2 + (off - (unsigned)NROWS * DDIM));
  v4h h;
  h[0] = (_Float16)v.x; h[1] = (_Float16)v.y;
  h[2] = (_Float16)v.z; h[3] = (_Float16)v.w;
  *(v4h*)(zh + off) = h;
  if (g < TWON) rowsum[g] = 0.0f;
  if (g == 0)   out[0]   = 0.0f;
}

// ---------------------------------------------------------------------------
// pos: exact fp32 dot(z1[i], z2[i]) / TEMP, one wave per row
// ---------------------------------------------------------------------------
__global__ __launch_bounds__(256) void pos_kernel(const float* __restrict__ z1,
                                                  const float* __restrict__ z2,
                                                  float* __restrict__ pos) {
  int row  = blockIdx.x * 8 + (threadIdx.x >> 5);
  int lane = threadIdx.x & 31;
  const float* a = z1 + (size_t)row * DDIM;
  const float* b = z2 + (size_t)row * DDIM;
  float s = 0.0f;
#pragma unroll
  for (int k = lane * 4; k < DDIM; k += 128) {
    float4 x = *(const float4*)(a + k);
    float4 y = *(const float4*)(b + k);
    s = fmaf(x.x, y.x, s); s = fmaf(x.y, y.y, s);
    s = fmaf(x.z, y.z, s); s = fmaf(x.w, y.w, s);
  }
  s += __shfl_xor(s, 16, 32);
  s += __shfl_xor(s, 8, 32);
  s += __shfl_xor(s, 4, 32);
  s += __shfl_xor(s, 2, 32);
  s += __shfl_xor(s, 1, 32);
  if (lane == 0) pos[row] = s * INVT;
}

// ---------------------------------------------------------------------------
// simlse: fused Z.Z^T (f16 WMMA, fp32 acc) + fixed-shift exp-sum per row.
// The WG's full 128x1024 A-strip stays RESIDENT in LDS (staged once, async);
// only B chunks stream through a double buffer via async loads.
// LDS: A 258KB + B 2x17KB = 292KB (<= 320KB/WG per CDNA5 ISA).
// ---------------------------------------------------------------------------
__global__ __launch_bounds__(256) void simlse_kernel(const _Float16* __restrict__ zh,
                                                     float* __restrict__ rowsum) {
  __shared__ _Float16 sA[A_TOT];          // resident A strip
  __shared__ _Float16 sB[2 * B_HALVES];   // double-buffered B chunks

  const int tid   = threadIdx.x;
  const int lane  = tid & 31;
  const int w     = tid >> 5;       // wave id 0..7
  const int nn    = lane & 15;      // row (A) / col (B) within tile
  const int hi16  = lane >> 4;      // lane-half selects K sub-range
  const int rowbase = blockIdx.x * ROWS_WG;
  const int jstart  = blockIdx.y * JBLKS;

  // ---- stage resident A strip: 128 rows x 1024 halves, 64 b128 per thread
#pragma unroll 4
  for (int it = 0; it < 64; ++it) {
    int c   = tid + it * 256;
    int row = c >> 7;               // 128 b128-chunks per row
    int k8  = (c & 127) * 8;
    async_cp16(zh + (size_t)(rowbase + row) * DDIM + k8, sA + row * ASTR + k8);
  }

  // ---- stage one B chunk: 64 rows x 128 halves, 4 b128 per thread
  auto stageB = [&](int colb, int kc, int buf) {
    _Float16* dB = sB + buf * B_HALVES;
#pragma unroll
    for (int it = 0; it < 4; ++it) {
      int c   = tid + it * 256;
      int row = c >> 4;             // 16 b128-chunks per row
      int k8  = (c & 15) * 8;
      async_cp16(zh + (size_t)(colb + row) * DDIM + kc + k8, dB + row * BST + k8);
    }
  };

  float rs[8];
#pragma unroll
  for (int r = 0; r < 8; ++r) rs[r] = 0.0f;

  stageB(jstart * COLS_BLK, 0, 0);  // prologue: B chunk 0 -> buffer 0
  int cc = 0;

  for (int jb = jstart; jb < jstart + JBLKS; ++jb) {
    const int colb = jb * COLS_BLK;
    v8f acc[4] = { {0,0,0,0,0,0,0,0}, {0,0,0,0,0,0,0,0},
                   {0,0,0,0,0,0,0,0}, {0,0,0,0,0,0,0,0} };

    for (int kci = 0; kci < KCHUNKS; ++kci) {
      const int buf   = cc & 1;
      const int kbase = kci * KCH;
      const int ncc   = cc + 1;
      if (ncc < JBLKS * KCHUNKS) {
        const int njb = jstart + (ncc >> 3);          // KCHUNKS == 8
        const int nkc = (ncc & (KCHUNKS - 1)) * KCH;
        stageB(njb * COLS_BLK, nkc, ncc & 1);
        wait_async_le4();  // in-order: chunk cc (and A on first pass) complete
      } else {
        wait_async_le0();
      }
      __syncthreads();     // chunk cc visible to every wave

      const _Float16* bB = sB + buf * B_HALVES;
      const _Float16* arow = sA + (w * 16 + nn) * ASTR + kbase;

#pragma unroll
      for (int kk2 = 0; kk2 < KCH; kk2 += 64) {
        // batch fragment loads for two K-steps, then 8 WMMAs back-to-back
        v16h af[2];
        v16h bf[2][4];
#pragma unroll
        for (int s = 0; s < 2; ++s) {
          const int kk = kk2 + s * 32;
          v8h a0 = *(const v8h*)(arow + kk + 8 * hi16);
          v8h a1 = *(const v8h*)(arow + kk + 16 + 8 * hi16);
#pragma unroll
          for (int i = 0; i < 8; ++i) { af[s][i] = a0[i]; af[s][i + 8] = a1[i]; }
#pragma unroll
          for (int t = 0; t < 4; ++t) {
            const _Float16* br = bB + (t * 16 + nn) * BST + kk + 8 * hi16;
            v8h b0 = *(const v8h*)(br);
            v8h b1 = *(const v8h*)(br + 16);
#pragma unroll
            for (int i = 0; i < 8; ++i) { bf[s][t][i] = b0[i]; bf[s][t][i + 8] = b1[i]; }
          }
        }
#pragma unroll
        for (int s = 0; s < 2; ++s)
#pragma unroll
          for (int t = 0; t < 4; ++t)
            acc[t] = __builtin_amdgcn_wmma_f32_16x16x32_f16(
                false, af[s], false, bf[s][t], (short)0, acc[t], false, false);
      }

      __syncthreads();     // all waves done with buf before it is re-staged
      ++cc;
    }

    // epilogue: exp(sim - S) with diagonal masked, per-lane row partials.
#pragma unroll
    for (int t = 0; t < 4; ++t) {
      int gcol = colb + t * 16 + nn;
#pragma unroll
      for (int r = 0; r < 8; ++r) {
        int grow = rowbase + w * 16 + r + 8 * hi16;
        float e  = __expf((acc[t][r] - 1.0f) * INVT);  // exp(sim - 1/T)
        rs[r] += (grow == gcol) ? 0.0f : e;
      }
    }
  }

  // reduce across the 16 lanes sharing a row (xor 1,2,4,8 stays in-half)
#pragma unroll
  for (int r = 0; r < 8; ++r) {
    float v = rs[r];
    v += __shfl_xor(v, 1, 32);
    v += __shfl_xor(v, 2, 32);
    v += __shfl_xor(v, 4, 32);
    v += __shfl_xor(v, 8, 32);
    if (nn == 0)
      atomicAdd(&rowsum[rowbase + w * 16 + r + 8 * hi16], v);
  }
}

// ---------------------------------------------------------------------------
// finish: lse = log(rowsum) + S; loss = mean(lse - pos)
// ---------------------------------------------------------------------------
__global__ __launch_bounds__(256) void finish_kernel(const float* __restrict__ rowsum,
                                                     const float* __restrict__ pos,
                                                     float* __restrict__ out) {
  int g    = blockIdx.x * 256 + threadIdx.x;
  int lane = threadIdx.x & 31;
  float lse = __logf(rowsum[g]) + INVT;
  float p   = pos[g & (NROWS - 1)];
  float c   = (lse - p) * (1.0f / TWON);
  c += __shfl_xor(c, 16, 32);
  c += __shfl_xor(c, 8, 32);
  c += __shfl_xor(c, 4, 32);
  c += __shfl_xor(c, 2, 32);
  c += __shfl_xor(c, 1, 32);
  if (lane == 0) atomicAdd(out, c);
}

// ---------------------------------------------------------------------------
extern "C" void kernel_launch(void* const* d_in, const int* in_sizes, int n_in,
                              void* d_out, int out_size, void* d_ws, size_t ws_size,
                              hipStream_t stream) {
  (void)in_sizes; (void)n_in; (void)out_size; (void)ws_size;
  const float* z1 = (const float*)d_in[0];
  const float* z2 = (const float*)d_in[1];
  float* out = (float*)d_out;

  // workspace layout: pos[4096] f32 | rowsum[8192] f32 @16KB | zh[8192*1024] f16 @64KB
  float*    pos    = (float*)d_ws;
  float*    rowsum = (float*)((char*)d_ws + 16384);
  _Float16* zh     = (_Float16*)((char*)d_ws + 65536);

  prep_kernel  <<<TWON * DDIM / (256 * 4), 256, 0, stream>>>(z1, z2, zh, rowsum, out);
  pos_kernel   <<<NROWS / 8,               256, 0, stream>>>(z1, z2, pos);
  simlse_kernel<<<dim3(TWON / ROWS_WG, COL_SPLITS), 256, 0, stream>>>(zh, rowsum);
  finish_kernel<<<TWON / 256,              256, 0, stream>>>(rowsum, pos, out);
}
